// GNAN_44873818308787
// MI455X (gfx1250) — matrix operand order repaired and checked by reading, not compiled, not run on hardware
//
#include <hip/hip_runtime.h>
#include <hip/hip_bf16.h>

typedef __attribute__((ext_vector_type(16))) _Float16 v16h;
typedef __attribute__((ext_vector_type(8)))  float    v8f;
typedef __attribute__((ext_vector_type(2)))  float    v2f;

#define Nn 2048
#define Ff 64
#define Hh 16
#define Oo 8
#define JSPLIT 8
#define JCHUNK 32
#define IB 16
#define TS 17   // padded LDS tile stride (gcd(17,64)=1 -> conflict-free columns)

__global__ void gnan_zero_out(float* out, int n) {
    int i = blockIdx.x * blockDim.x + threadIdx.x;
    if (i < n) out[i] = 0.f;
}

// ---- f_sums: one 64-thread block per node; thread f runs its feature MLP ----
__global__ void gnan_fsums(const float* __restrict__ x,
                           const float* __restrict__ fW1, const float* __restrict__ fb1,
                           const float* __restrict__ fW2, const float* __restrict__ fb2,
                           const float* __restrict__ fW3, const float* __restrict__ fb3,
                           float* __restrict__ fsums) {
    __shared__ float red[Ff][Oo];
    const int n = blockIdx.x;
    const int f = threadIdx.x;          // 0..63
    const float xv = x[n * Ff + f];
    float h1[Hh];
#pragma unroll
    for (int h = 0; h < Hh; ++h) {
        float v = xv * fW1[f * Hh + h] + fb1[f * Hh + h];
        h1[h] = v > 0.f ? v : 0.f;
    }
    float h2[Hh];
#pragma unroll
    for (int g = 0; g < Hh; ++g) {
        float a = fb2[f * Hh + g];
#pragma unroll
        for (int h = 0; h < Hh; ++h) a += fW2[(f * Hh + g) * Hh + h] * h1[h];
        h2[g] = a > 0.f ? a : 0.f;
    }
#pragma unroll
    for (int o = 0; o < Oo; ++o) {
        float a = fb3[f * Oo + o];
#pragma unroll
        for (int h = 0; h < Hh; ++h) a += fW3[(f * Oo + o) * Hh + h] * h2[h];
        red[f][o] = a;
    }
    __syncthreads();
    if (f < Oo) {
        float s = 0.f;
#pragma unroll 8
        for (int ff = 0; ff < Ff; ++ff) s += red[ff][f];
        fsums[n * Oo + f] = s;
    }
}

// ---- fused m-MLP (f32 WMMA 16x16x4 matvec) + (m_norm @ f_sums) f16 WMMA ----
__global__ void __launch_bounds__(256)
gnan_main(const float* __restrict__ dist, const float* __restrict__ norm,
          const float* __restrict__ mW1, const float* __restrict__ mb1,
          const float* __restrict__ mW2, const float* __restrict__ mb2,
          const float* __restrict__ mW3, const float* __restrict__ mb3,
          const float* __restrict__ fsums, float* __restrict__ out) {
    __shared__ float sD[8][JCHUNK * TS];
    __shared__ float sNm[8][JCHUNK * TS];

    const int lane = threadIdx.x & 31;
    const int w    = threadIdx.x >> 5;
    const int gw   = blockIdx.x * 8 + w;
    const int iblk = gw / JSPLIT;
    const int js   = gw % JSPLIT;
    const int i0   = iblk * IB;
    const int jspan  = Nn / JSPLIT;      // 256 j per wave
    const int jstart = js * jspan;

    const int half = lane >> 4;          // 0 | 1
    const int lm16 = lane & 15;

    // f32 WMMA 16x16x4 operand layout: lane's two elements carry
    // K = 4q + 2*half + {0,1} for chunk q (A: M=lm16 row; B: N=lm16 col).
    // Per-lane layer-1 weights for exactly the 8 h-indices this lane feeds:
    float w1v[8], b1v[8];
    v2f aW2q[4];
#pragma unroll
    for (int q = 0; q < 4; ++q) {
#pragma unroll
        for (int e = 0; e < 2; ++e) {
            const int h = 4 * q + 2 * half + e;
            w1v[2 * q + e] = mW1[h];
            b1v[2 * q + e] = mb1[h];
            aW2q[q][e]     = mW2[lm16 * Hh + h];   // row g=lm16 of mW2
        }
    }
    float b2v[8], w3v[8];
#pragma unroll
    for (int r = 0; r < 8; ++r) {
        b2v[r] = mb2[half * 8 + r];
        w3v[r] = mW3[half * 8 + r];
    }
    const float b3 = mb3[0];

    v8f acc = {};  // pred tile [16 i x 16 o] f32 accumulator (persistent over j)

    for (int jc = 0; jc < jspan / JCHUNK; ++jc) {
        const int jb = jstart + jc * JCHUNK;

        // Stage d/norm tile [32 j][16 i] into LDS: coalesced over j, conflict-free columns
#pragma unroll
        for (int idx = 0; idx < IB * JCHUNK; idx += 32) {
            int t  = idx + lane;
            int ii = t >> 5;             // 0..15
            int jj = t & 31;             // 0..31
            sD[w][jj * TS + ii]  = dist[(size_t)(i0 + ii) * Nn + jb + jj];
            sNm[w][jj * TS + ii] = norm[(size_t)(i0 + ii) * Nn + jb + jj];
        }
        __syncthreads();

        float mn[JCHUNK];
#pragma unroll
        for (int jj = 0; jj < JCHUNK; ++jj) {
            const float dv = sD[w][jj * TS + lm16];
            // Layer-2 matvec as 4 chained f32 WMMAs over K=16 (full utilization):
            // D[16 g x 16 i-pairs] = mW2[16x16] x hm1[16x16]
            v8f h2 = {};
#pragma unroll
            for (int q = 0; q < 4; ++q) {
                v2f bq;
                float v0 = dv * w1v[2 * q]     + b1v[2 * q];
                float v1 = dv * w1v[2 * q + 1] + b1v[2 * q + 1];
                bq[0] = v0 > 0.f ? v0 : 0.f;
                bq[1] = v1 > 0.f ? v1 : 0.f;
                h2 = __builtin_amdgcn_wmma_f32_16x16x4_f32(false, aW2q[q], false, bq,
                                                           (short)0, h2, false, false);
            }
            float part = 0.f;
#pragma unroll
            for (int r = 0; r < 8; ++r) {
                float t = h2[r] + b2v[r];
                t = t > 0.f ? t : 0.f;
                part += t * w3v[r];
            }
            const float tot = part + __shfl_xor(part, 16, 32);   // fold g halves
            mn[jj] = (tot + b3) / sNm[w][jj * TS + lm16];        // m_norm[i=lm16][jb+jj]
        }
        __syncthreads();

        // A operand: m_norm [16 i x 32 j] f16 (row i = lm16 lives in this lane)
        v16h aM;
#pragma unroll
        for (int e = 0; e < 8; ++e) {
            float lo = half ? mn[e + 8]  : mn[e];        // K = e + 8*half
            float hi = half ? mn[e + 24] : mn[e + 16];   // K = e + 16 + 8*half
            aM[e]     = (_Float16)lo;
            aM[e + 8] = (_Float16)hi;
        }
        // B operand: f_sums [32 j x 16 o] f16, columns o>=8 zero
        v16h bF;
        const int kb = jb + half * 8;
#pragma unroll
        for (int e = 0; e < 16; ++e) {
            float v = fsums[(size_t)(kb + e + (e >= 8 ? 8 : 0)) * Oo + (lm16 < Oo ? lm16 : 0)];
            bF[e] = (lm16 < Oo) ? (_Float16)v : (_Float16)0.0f;
        }
        acc = __builtin_amdgcn_wmma_f32_16x16x32_f16(false, aM, false, bF,
                                                     (short)0, acc, false, false);
    }

    // Merge partial pred tiles across j-splits
    if (lm16 < Oo) {
#pragma unroll
        for (int r = 0; r < 8; ++r) {
            int i = i0 + half * 8 + r;
            atomicAdd(&out[i * Oo + lm16], acc[r]);
        }
    }
}

extern "C" void kernel_launch(void* const* d_in, const int* in_sizes, int n_in,
                              void* d_out, int out_size, void* d_ws, size_t ws_size,
                              hipStream_t stream) {
    (void)in_sizes; (void)n_in; (void)out_size; (void)ws_size;
    const float* x    = (const float*)d_in[0];
    const float* dist = (const float*)d_in[1];
    const float* nrm  = (const float*)d_in[2];
    const float* fW1  = (const float*)d_in[3];
    const float* fb1  = (const float*)d_in[4];
    const float* fW2  = (const float*)d_in[5];
    const float* fb2  = (const float*)d_in[6];
    const float* fW3  = (const float*)d_in[7];
    const float* fb3  = (const float*)d_in[8];
    const float* mW1  = (const float*)d_in[9];
    const float* mb1  = (const float*)d_in[10];
    const float* mW2  = (const float*)d_in[11];
    const float* mb2  = (const float*)d_in[12];
    const float* mW3  = (const float*)d_in[13];
    const float* mb3  = (const float*)d_in[14];
    float* out   = (float*)d_out;
    float* fsums = (float*)d_ws;   // 2048*8 f32 = 64 KB scratch

    gnan_zero_out<<<(Nn * Oo + 255) / 256, 256, 0, stream>>>(out, Nn * Oo);
    gnan_fsums<<<Nn, Ff, 0, stream>>>(x, fW1, fb1, fW2, fb2, fW3, fb3, fsums);
    const int totalWaves = (Nn / IB) * JSPLIT;   // 1024 waves
    gnan_main<<<totalWaves / 8, 256, 0, stream>>>(dist, nrm, mW1, mb1, mW2, mb2,
                                                  mW3, mb3, fsums, out);
}
